// FeatureExtaction_81028853007031
// MI455X (gfx1250) — matrix operand order repaired and checked by reading, not compiled, not run on hardware
//
#include <hip/hip_runtime.h>

// ---------------- problem constants ----------------
#define EMB     300
#define HID     150
#define GATES   600      // 4*HID
#define BATCH   128
#define TSTEPS  512
#define NPAD    608      // GATES padded to 16
#define KPAD_E  320      // EMB padded to 32
#define KPAD_H  160      // HID padded to 32
#define MROWS   65536    // BATCH*TSTEPS
#define NTILES  38       // NPAD/16

typedef __attribute__((ext_vector_type(16))) __bf16 v16bf;
typedef __attribute__((ext_vector_type(8)))  float  v8f;

union AF {
    uint4          q[2];
    unsigned short us[16];
    v16bf          v;
};

__device__ __forceinline__ unsigned short f2bf_bits(float f) {
    union { float f; unsigned u; } v; v.f = f;
    unsigned r = v.u + 0x7FFFu + ((v.u >> 16) & 1u);   // round-to-nearest-even
    return (unsigned short)(r >> 16);
}
__device__ __forceinline__ float bf2f(unsigned short h) {
    union { unsigned u; float f; } v; v.u = ((unsigned)h) << 16;
    return v.f;
}
__device__ __forceinline__ float fast_sig(float x)  { return 1.0f / (1.0f + __expf(-x)); }
__device__ __forceinline__ float fast_tanh(float x) { return 1.0f - 2.0f / (__expf(2.0f * x) + 1.0f); }

// CDNA5 async global->LDS copy (16 B per active lane, tracked by ASYNCcnt).
// lds_off = byte offset relative to workgroup LDS base (dynamic LDS starts at 0 here).
__device__ __forceinline__ void async_ld16(unsigned lds_off, const void* g) {
    unsigned long long ga = (unsigned long long)g;
    asm volatile("global_load_async_to_lds_b128 %0, %1, off"
                 :: "v"(lds_off), "v"(ga) : "memory");
}
__device__ __forceinline__ void wait_async0() {
    asm volatile("s_wait_asynccnt 0x0" ::: "memory");
}

// ---------------- kernel 0a: pack weights to bf16 (padded) + bias sums ----------
__global__ void pack_kernel(const float* __restrict__ w_ih_f, const float* __restrict__ w_hh_f,
                            const float* __restrict__ b_ih_f, const float* __restrict__ b_hh_f,
                            const float* __restrict__ w_ih_b, const float* __restrict__ w_hh_b,
                            const float* __restrict__ b_ih_b, const float* __restrict__ b_hh_b,
                            unsigned short* __restrict__ wih_pack,   // [2][NPAD][KPAD_E]
                            unsigned short* __restrict__ whh_pack,   // [2][NPAD][KPAD_H]
                            float* __restrict__ bias_sum)            // [2][GATES]
{
    const int NW = 2 * NPAD * KPAD_E;
    const int NH = 2 * NPAD * KPAD_H;
    int idx = blockIdx.x * 256 + threadIdx.x;
    if (idx < NW) {
        int dir = idx / (NPAD * KPAD_E);
        int rem = idx % (NPAD * KPAD_E);
        int n = rem / KPAD_E, k = rem % KPAD_E;
        const float* w = dir ? w_ih_b : w_ih_f;
        float v = (n < GATES && k < EMB) ? w[n * EMB + k] : 0.0f;
        wih_pack[idx] = f2bf_bits(v);
    } else if (idx < NW + NH) {
        int i2 = idx - NW;
        int dir = i2 / (NPAD * KPAD_H);
        int rem = i2 % (NPAD * KPAD_H);
        int n = rem / KPAD_H, k = rem % KPAD_H;
        const float* w = dir ? w_hh_b : w_hh_f;
        float v = (n < GATES && k < HID) ? w[n * HID + k] : 0.0f;
        whh_pack[i2] = f2bf_bits(v);
    } else if (idx < NW + NH + 2 * GATES) {
        int i3 = idx - NW - NH;
        int dir = i3 / GATES, g = i3 % GATES;
        bias_sum[i3] = dir ? (b_ih_b[g] + b_hh_b[g]) : (b_ih_f[g] + b_hh_f[g]);
    }
}

// ---------------- kernel 0b: pack x to bf16 once (padded K) ----------------------
__global__ __launch_bounds__(256) void pack_x_kernel(const float* __restrict__ x,
                                                     unsigned short* __restrict__ x_bf)
{
    int idx = blockIdx.x * 256 + threadIdx.x;                 // over MROWS*KPAD_E
    if (idx < MROWS * KPAD_E) {
        int row = idx / KPAD_E, k = idx % KPAD_E;
        float v = (k < EMB) ? x[(size_t)row * EMB + k] : 0.0f;
        x_bf[idx] = f2bf_bits(v);
    }
}

// ---------------- kernel 1: input projection GEMM -------------------------------
// xg[dir][(b*T+t)][g] (bf16) = x[(b*T+t),:] @ w_ih[dir]^T + (b_ih+b_hh)
// grid: (512 M-blocks of 128 rows, 2 dirs), 256 threads = 8 waves.
// LDS: x tile 128x320 bf16 (80 KB, async-staged once)
//    + double-buffered 16-col w slabs (2 x 10 KB, async-streamed).
__global__ __launch_bounds__(256) void xproj_kernel(const unsigned short* __restrict__ x_bf,
                                                    const unsigned short* __restrict__ wih_pack,
                                                    const float* __restrict__ bias_sum,
                                                    unsigned short* __restrict__ xg)
{
    extern __shared__ char smem[];
    unsigned short* xls = (unsigned short*)smem;                        // [128][KPAD_E]
    unsigned short* bls = (unsigned short*)(smem + 128 * KPAD_E * 2);   // 2 x [16][KPAD_E]
    const unsigned BOFF = 128 * KPAD_E * 2;              // LDS byte offset of slab buffers
    const unsigned SLAB = 16 * KPAD_E * 2;               // 10,240 B per slab

    const int tid  = threadIdx.x;
    const int lane = tid & 31, wave = tid >> 5;
    const int lm   = lane & 15, hi = lane >> 4;
    const int mb   = blockIdx.x * 128;      // row block
    const int dir  = blockIdx.y;

    const unsigned short* wbase = wih_pack + (size_t)dir * NPAD * KPAD_E;

    // async-stage x tile and w slab 0
    const unsigned short* xrow = x_bf + (size_t)mb * KPAD_E;
    for (int i = tid; i < 128 * KPAD_E / 8; i += 256)
        async_ld16((unsigned)(i * 16), xrow + i * 8);
    for (int i = tid; i < 16 * KPAD_E / 8; i += 256)
        async_ld16(BOFF + (unsigned)(i * 16), wbase + i * 8);
    wait_async0();
    __syncthreads();

    // preload all 10 A fragments for this wave's 16 rows
    AF af[10];
    #pragma unroll
    for (int s = 0; s < 10; ++s) {
        const int kb = s * 32 + hi * 8;
        af[s].q[0] = *(const uint4*)(xls + (wave * 16 + lm) * KPAD_E + kb);
        af[s].q[1] = *(const uint4*)(xls + (wave * 16 + lm) * KPAD_E + kb + 16);
    }

    unsigned short* xo = xg + (size_t)dir * MROWS * GATES;

    for (int j = 0; j < NTILES; ++j) {
        // stream w slab j+1 into the other buffer while computing slab j
        if (j + 1 < NTILES) {
            const unsigned short* wp = wbase + (size_t)(j + 1) * 16 * KPAD_E;
            const unsigned boff = BOFF + (unsigned)((j + 1) & 1) * SLAB;
            for (int i = tid; i < 16 * KPAD_E / 8; i += 256)
                async_ld16(boff + (unsigned)(i * 16), wp + i * 8);
        }

        const unsigned short* bcur = bls + (j & 1) * (16 * KPAD_E);

        // software-pipelined B fragments: ds loads for s+1 overlap wmma for s
        AF bb[2];
        bb[0].q[0] = *(const uint4*)(bcur + lm * KPAD_E + hi * 8);
        bb[0].q[1] = *(const uint4*)(bcur + lm * KPAD_E + hi * 8 + 16);
        v8f acc = {0.f, 0.f, 0.f, 0.f, 0.f, 0.f, 0.f, 0.f};
        #pragma unroll
        for (int s = 0; s < 10; ++s) {
            if (s + 1 < 10) {
                const int kb = (s + 1) * 32 + hi * 8;
                bb[(s + 1) & 1].q[0] = *(const uint4*)(bcur + lm * KPAD_E + kb);
                bb[(s + 1) & 1].q[1] = *(const uint4*)(bcur + lm * KPAD_E + kb + 16);
            }
            acc = __builtin_amdgcn_wmma_f32_16x16x32_bf16(false, af[s].v, false, bb[s & 1].v,
                                                          (short)0, acc, false, false);
        }

        const int n = j * 16 + lm;
        if (n < GATES) {
            const float bsum = bias_sum[dir * GATES + n];
            #pragma unroll
            for (int r = 0; r < 8; ++r) {
                int rr = mb + wave * 16 + r + 8 * hi;
                xo[(size_t)rr * GATES + n] = f2bf_bits(acc[r] + bsum);
            }
        }
        wait_async0();       // slab j+1 fully in LDS
        __syncthreads();
    }
}

// ---------------- kernel 2: serial LSTM scan (both directions) ------------------
// grid: (8 batch-slices of 16, 2 dirs), 256 threads = 8 waves.
// LDS: w_hh^T bf16 [608][160] (194.5 KB) + gates f32 [16][608] (38.9 KB) + h bf16 [16][160]
__global__ __launch_bounds__(256) void lstm_scan_kernel(const unsigned short* __restrict__ whh_pack,
                                                        const unsigned short* __restrict__ xg,
                                                        const int* __restrict__ lens,
                                                        float* __restrict__ out)
{
    extern __shared__ char smem[];
    unsigned short* wls = (unsigned short*)smem;                                        // [608][160]
    float*          gls = (float*)(smem + NPAD * KPAD_H * 2);                           // [16][608]
    unsigned short* hls = (unsigned short*)(smem + NPAD * KPAD_H * 2 + 16 * NPAD * 4);  // [16][160]
    const unsigned HOFF = NPAD * KPAD_H * 2 + 16 * NPAD * 4;

    const int tid  = threadIdx.x;
    const int dir  = blockIdx.y;
    const int b0   = blockIdx.x * 16;
    const int lane = tid & 31, wave = tid >> 5;
    const int lm   = lane & 15, hi = lane >> 4;

    const unsigned short* whh = whh_pack + (size_t)dir * NPAD * KPAD_H;
    const unsigned short* xgd = xg       + (size_t)dir * MROWS * GATES;

    // async-stage w_hh^T into LDS (resident for all 512 steps)
    for (int i = tid; i < NPAD * KPAD_H / 8; i += 256)
        async_ld16((unsigned)(i * 16), whh + i * 8);
    // zero h (covers K padding 150..159 as well)
    for (int i = tid; i < 16 * KPAD_H / 2; i += 256)
        ((unsigned*)hls)[i] = 0u;
    (void)HOFF;

    // per-thread elementwise assignment: e = tid + 256*k over 16*150 = 2400 (b,j) pairs
    float cst[10], macc[10], hk[10];
    int   eb[10], ej[10], el[10];
    #pragma unroll
    for (int k = 0; k < 10; ++k) {
        int e = tid + 256 * k;
        bool ok = e < 16 * HID;
        eb[k] = ok ? e / HID : 0;
        ej[k] = ok ? e % HID : 0;
        el[k] = ok ? lens[b0 + eb[k]] : 1;
        cst[k] = 0.f; macc[k] = 0.f; hk[k] = 0.f;
    }
    wait_async0();
    __syncthreads();

    const int ntiles = (wave < 6) ? 5 : 4;    // 38 N-tiles split over 8 waves

    for (int step = 0; step < TSTEPS; ++step) {
        const int t = dir ? (TSTEPS - 1 - step) : step;

        // A fragments: h in LDS, row = batch-lane, K packed per documented bf16 layout
        AF af[5];
        #pragma unroll
        for (int s = 0; s < 5; ++s) {
            const int kb = s * 32 + hi * 8;
            af[s].q[0] = *(const uint4*)(hls + lm * KPAD_H + kb);
            af[s].q[1] = *(const uint4*)(hls + lm * KPAD_H + kb + 16);
        }

        for (int q = 0; q < ntiles; ++q) {
            const int j = wave + 8 * q;
            const unsigned short* wrow = wls + (j * 16 + lm) * KPAD_H;

            AF bb[2];
            bb[0].q[0] = *(const uint4*)(wrow + hi * 8);
            bb[0].q[1] = *(const uint4*)(wrow + hi * 8 + 16);
            v8f acc = {0.f, 0.f, 0.f, 0.f, 0.f, 0.f, 0.f, 0.f};
            #pragma unroll
            for (int s = 0; s < 5; ++s) {
                if (s + 1 < 5) {
                    const int kb = (s + 1) * 32 + hi * 8;
                    bb[(s + 1) & 1].q[0] = *(const uint4*)(wrow + kb);
                    bb[(s + 1) & 1].q[1] = *(const uint4*)(wrow + kb + 16);
                }
                acc = __builtin_amdgcn_wmma_f32_16x16x32_bf16(false, af[s].v, false, bb[s & 1].v,
                                                              (short)0, acc, false, false);
            }
            #pragma unroll
            for (int r = 0; r < 8; ++r)
                gls[(r + 8 * hi) * NPAD + j * 16 + lm] = acc[r];
        }
        __syncthreads();

        // elementwise LSTM cell update
        #pragma unroll
        for (int k = 0; k < 10; ++k) {
            int e = tid + 256 * k;
            if (e < 16 * HID) {
                const int b = eb[k], j = ej[k];
                const size_t xoff = ((size_t)(b0 + b) * TSTEPS + t) * GATES;
                float gi = gls[b * NPAD + j]         + bf2f(xgd[xoff + j]);
                float gf = gls[b * NPAD + HID + j]   + bf2f(xgd[xoff + HID + j]);
                float gg = gls[b * NPAD + 2*HID + j] + bf2f(xgd[xoff + 2*HID + j]);
                float go = gls[b * NPAD + 3*HID + j] + bf2f(xgd[xoff + 3*HID + j]);
                float si = fast_sig(gi), sf = fast_sig(gf), so = fast_sig(go);
                float c  = sf * cst[k] + si * fast_tanh(gg);
                cst[k]   = c;
                float h  = so * fast_tanh(c);
                hls[b * KPAD_H + j] = f2bf_bits(h);
                if (t <  el[k])     macc[k] += h;
                if (t == el[k] - 1) hk[k]   = h;
            }
        }
        __syncthreads();
    }

    // epilogue: tanh(concat([mean, h_k]))
    #pragma unroll
    for (int k = 0; k < 10; ++k) {
        int e = tid + 256 * k;
        if (e < 16 * HID) {
            const int b = eb[k], j = ej[k];
            const int row = (b0 + b) * GATES;                 // out is [B, 4H=600]
            const int cm  = dir ? (HID + j) : j;              // mean cols
            const int ch  = dir ? (3*HID + j) : (2*HID + j);  // h_k cols
            out[row + cm] = fast_tanh(macc[k] / (float)el[k]);
            out[row + ch] = fast_tanh(hk[k]);
        }
    }
}

// ---------------- host side ------------------------------------------------------
extern "C" void kernel_launch(void* const* d_in, const int* in_sizes, int n_in,
                              void* d_out, int out_size, void* d_ws, size_t ws_size,
                              hipStream_t stream) {
    (void)in_sizes; (void)n_in; (void)out_size; (void)ws_size;
    const float* x      = (const float*)d_in[0];
    const float* w_ih_f = (const float*)d_in[1];
    const float* w_hh_f = (const float*)d_in[2];
    const float* b_ih_f = (const float*)d_in[3];
    const float* b_hh_f = (const float*)d_in[4];
    const float* w_ih_b = (const float*)d_in[5];
    const float* w_hh_b = (const float*)d_in[6];
    const float* b_ih_b = (const float*)d_in[7];
    const float* b_hh_b = (const float*)d_in[8];
    const int*   lens   = (const int*)d_in[9];
    float* out = (float*)d_out;

    // workspace layout (256B aligned)
    char* ws = (char*)d_ws;
    unsigned short* wih_pack = (unsigned short*)(ws + 0);          //    778,240 B
    unsigned short* whh_pack = (unsigned short*)(ws + 778240);     //    389,120 B
    float*          bias_sum = (float*)(ws + 1167360);             //      4,800 B
    unsigned short* x_bf     = (unsigned short*)(ws + 1172224);    // 41,943,040 B
    unsigned short* xg       = (unsigned short*)(ws + 43115264);   // 157,286,400 B  (~200.4 MB)

    const int total_pack = 2 * NPAD * KPAD_E + 2 * NPAD * KPAD_H + 2 * GATES;
    pack_kernel<<<(total_pack + 255) / 256, 256, 0, stream>>>(
        w_ih_f, w_hh_f, b_ih_f, b_hh_f, w_ih_b, w_hh_b, b_ih_b, b_hh_b,
        wih_pack, whh_pack, bias_sum);

    pack_x_kernel<<<(MROWS * KPAD_E + 255) / 256, 256, 0, stream>>>(x, x_bf);

    const size_t smem_xp = (size_t)(128 + 2 * 16) * KPAD_E * 2;   // 102,400 B
    (void)hipFuncSetAttribute((const void*)xproj_kernel,
                              hipFuncAttributeMaxDynamicSharedMemorySize, (int)smem_xp);
    xproj_kernel<<<dim3(MROWS / 128, 2), 256, smem_xp, stream>>>(x_bf, wih_pack, bias_sum, xg);

    const size_t smem_sc = (size_t)NPAD * KPAD_H * 2   // w_hh^T bf16
                         + (size_t)16 * NPAD * 4       // gates f32
                         + (size_t)16 * KPAD_H * 2;    // h bf16  => 238,592 B
    (void)hipFuncSetAttribute((const void*)lstm_scan_kernel,
                              hipFuncAttributeMaxDynamicSharedMemorySize, (int)smem_sc);
    lstm_scan_kernel<<<dim3(BATCH / 16, 2), 256, smem_sc, stream>>>(whh_pack, xg, lens, out);
}
